// GraphAttentionLayer_28071906247347
// MI455X (gfx1250) — matrix operand order repaired and checked by reading, not compiled, not run on hardware
//
#include <hip/hip_runtime.h>

// ---------------------------------------------------------------------------
// GAT layer for MI455X (gfx1250, wave32, WMMA, TDM).
//   K1: h = x @ W  (bf16 WMMA, f32 accum) -> h_f32 [8192,256], hT_bf16 [256,8192]
//   K2: s_src = h@a1, s_dst = h@a2  (wave reductions)
//   K3: fused flash-style masked softmax-attention:
//       e = mish(s_src[i]+s_dst[j]) masked by adj, online softmax,
//       acc += P @ h via WMMA, out = mish(acc / l).
// adj (256 MB) streamed exactly once (~11us @ 23.3 TB/s = the roofline floor).
// hT (4 MB) is L2-resident; each 32-col tile is DMA'd into LDS by the Tensor
// Data Mover (double-buffered, s_wait_tensorcnt pipelined) -- no VALU staging.
// ---------------------------------------------------------------------------

typedef __attribute__((ext_vector_type(16))) __bf16        v16bf;
typedef __attribute__((ext_vector_type(8)))  float         v8f;
typedef __attribute__((ext_vector_type(4)))  float         v4f;
typedef __attribute__((ext_vector_type(4)))  int           v4i;
typedef __attribute__((ext_vector_type(8)))  int           v8i;
typedef __attribute__((ext_vector_type(4)))  unsigned int  v4u;

#define N_NODES 8192
#define FIN     512
#define FOUT    256
#define NEGB    (-9.0e15f)
#define NK      (N_NODES / 32)

#if __has_builtin(__builtin_amdgcn_tensor_load_to_lds)
#define HAVE_TDM 1
#else
#define HAVE_TDM 0
#endif

// mish(x) = x * tanh(softplus(x)) = x * ((1+e^x)^2 - 1) / ((1+e^x)^2 + 1)
// -> one v_exp_f32 + one v_rcp_f32. Clamp exp input so t2 stays finite.
__device__ __forceinline__ float mishf(float x) {
  float ex = __expf(fminf(x, 30.0f));
  float t  = 1.0f + ex;
  float t2 = t * t;
  return x * (t2 - 1.0f) * __builtin_amdgcn_rcpf(t2 + 1.0f);
}

#if HAVE_TDM
// Issue a TDM load of a [256 x 32] bf16 tile of hT ([256][8192], row-major)
// starting at column j0, into LDS at byte offset lds_off (row-major [256][32]).
// D# per CDNA5 ISA ch.8: group0 = {count, lds_addr, global_addr, type=2},
// group1 = {data_size=2B, tensor_dim0=8192, tensor_dim1=256, tile_dim0=32,
//           tile_dim1=256, tensor_dim0_stride=8192}. 2D tile -> groups 2/3 = 0.
__device__ __forceinline__ void tdm_load_tile(const __bf16* gsrc, unsigned lds_off) {
  unsigned long long ga = (unsigned long long)gsrc;
  v4u g0;
  g0[0] = 1u;                                   // count=1, user mode, no gather
  g0[1] = lds_off;                              // lds_addr (bytes)
  g0[2] = (unsigned)ga;                         // global_addr[31:0]
  g0[3] = (unsigned)((ga >> 32) & 0x01FFFFFFull) | (2u << 30);  // [56:32] | type=2
  v8i g1;
  g1[0] = 0x10000;                              // data_size code 1 (2 bytes)
  g1[1] = (int)(8192u << 16);                   // tensor_dim0 = 8192 (bits 79:48)
  g1[2] = (int)(256u  << 16);                   // tensor_dim1 = 256  (bits 111:80)
  g1[3] = (int)(32u   << 16);                   // tile_dim0   = 32   (bits 127:112)
  g1[4] = 256;                                  // tile_dim1   = 256  (bits 143:128)
  g1[5] = 8192;                                 // tensor_dim0_stride (bits 207:160)
  g1[6] = 0;
  g1[7] = 0;
  v4i z4 = {};
#if defined(__clang_major__) && __clang_major__ >= 23
  v8i z8 = {};
  __builtin_amdgcn_tensor_load_to_lds(g0, g1, z4, z4, z8, 0);
#else
  __builtin_amdgcn_tensor_load_to_lds(g0, g1, z4, z4, 0);
#endif
}
#endif

// ---------------------------------------------------------------------------
// Kernel 1: h = x @ W.  One 16x16 output tile per wave, K-loop in steps of 32.
// Writes h (f32 row-major) and hT (bf16, transposed [FOUT][N]) for K3's TDM.
// ---------------------------------------------------------------------------
__global__ __launch_bounds__(256)
void gat_gemm1(const float* __restrict__ x, const float* __restrict__ w,
               float* __restrict__ h, __bf16* __restrict__ hT) {
  const int lane = threadIdx.x & 31;
  const int wave = threadIdx.x >> 5;
  const int tile = blockIdx.x * 8 + wave;      // 512 m-tiles x 16 n-tiles
  const int m0   = (tile >> 4) << 4;
  const int n0   = (tile & 15) << 4;
  const int r    = lane & 15;
  const int hi   = lane >> 4;                  // K-half select
  const int selA = hi ? 8 : 0;                 // A: lanes>=16 hold K 8..15/24..31
  const int selB = hi ? 16 : 0;                // unused; B uses selA (K 8..15/24..31)
  const int rsel = hi ? 8 : 0;
  const int col  = n0 + r;

  v8f c = {};
  for (int ks = 0; ks < FIN / 32; ++ks) {
    const int kb = ks * 32;
    // A fragment: x[m0+r, kb + {selA..selA+7, 16+selA..16+selA+7}]
    const float* ax = x + (size_t)(m0 + r) * FIN + kb + selA;
    v4f a0 = *(const v4f*)(ax +  0);
    v4f a1 = *(const v4f*)(ax +  4);
    v4f a2 = *(const v4f*)(ax + 16);
    v4f a3 = *(const v4f*)(ax + 20);
    v16bf A;
#pragma unroll
    for (int q = 0; q < 4; ++q) {
      A[q +  0] = (__bf16)a0[q];
      A[q +  4] = (__bf16)a1[q];
      A[q +  8] = (__bf16)a2[q];
      A[q + 12] = (__bf16)a3[q];
    }
    // B fragment: w[kb + {selA..selA+7, 16+selA..16+selA+7}, col] (strided column gather)
    const float* bw = w + (size_t)(kb + selA) * FOUT + col;
    v16bf B;
#pragma unroll
    for (int t = 0; t < 16; ++t) B[t] = (__bf16)bw[(t < 8 ? t : t + 8) * FOUT];

    c = __builtin_amdgcn_wmma_f32_16x16x32_bf16(false, A, false, B,
                                                (short)0, c, false, false);
  }
  // C layout: VGPR vr -> row m0+vr+rsel, col = n0 + (lane&15)
  union { __bf16 b[8]; v4i i; } pk;
#pragma unroll
  for (int vr = 0; vr < 8; ++vr) {
    const int row = m0 + vr + rsel;
    h[(size_t)row * FOUT + col] = c[vr];
    pk.b[vr] = (__bf16)c[vr];
  }
  // 8 consecutive rows of one hT column -> one b128 store
  *(v4i*)(hT + (size_t)col * N_NODES + m0 + rsel) = pk.i;
}

// ---------------------------------------------------------------------------
// Kernel 2: s_src[i] = h[i,:] . a[:256],  s_dst[i] = h[i,:] . a[256:512]
// ---------------------------------------------------------------------------
__global__ __launch_bounds__(256)
void gat_scores(const float* __restrict__ h, const float* __restrict__ a,
                float* __restrict__ s_src, float* __restrict__ s_dst) {
  const int lane = threadIdx.x & 31;
  const int row  = blockIdx.x * 8 + (threadIdx.x >> 5);
  float acc1 = 0.f, acc2 = 0.f;
#pragma unroll
  for (int t = 0; t < FOUT / 32; ++t) {
    const int k = lane + 32 * t;
    const float hv = h[(size_t)row * FOUT + k];
    acc1 = fmaf(hv, a[k], acc1);
    acc2 = fmaf(hv, a[FOUT + k], acc2);
  }
#pragma unroll
  for (int off = 16; off > 0; off >>= 1) {
    acc1 += __shfl_xor(acc1, off, 32);
    acc2 += __shfl_xor(acc2, off, 32);
  }
  if (lane == 0) { s_src[row] = acc1; s_dst[row] = acc2; }
}

// ---------------------------------------------------------------------------
// Kernel 3: fused masked-softmax-attention (flash style, online softmax).
// Block = 128 threads (4 waves), wave owns 16 rows, block owns 64 rows,
// grid = 128 blocks. K-loop over 32 adjacency columns per step.
// h-tile staged in LDS by TDM (double-buffered) from transposed hT.
// ---------------------------------------------------------------------------
__global__ __launch_bounds__(128)
void gat_flash(const int* __restrict__ adj, const __bf16* __restrict__ hT,
               const float* __restrict__ s_src, const float* __restrict__ s_dst,
               float* __restrict__ out) {
#if HAVE_TDM
  __shared__ __align__(16) __bf16 hbT[2][FOUT][32];   // 2 x 16 KB (double buffer)
#else
  __shared__ __align__(16) __bf16 hbT[1][FOUT][32];
#endif
  __shared__ float sdst[N_NODES];                     // 32 KB, block-resident

  const int tid  = threadIdx.x;
  const int lane = tid & 31;
  const int wave = tid >> 5;
  const int hi   = lane >> 4;
  const int r    = lane & 15;
  const int selA = hi ? 8 : 0;          // A-fragment K-half
  const int sel2 = hi ? 16 : 0;         // unused; B fragment uses selA
  const int rsel = hi ? 8 : 0;          // C rows: vr + rsel
  const int rowBase = blockIdx.x * 64 + wave * 16;
  const int myrow   = rowBase + r;
  const float ssrc  = s_src[myrow];

  for (int i = tid; i < N_NODES; i += 128) sdst[i] = s_dst[i];

  v8f acc[16];
#pragma unroll
  for (int nt = 0; nt < 16; ++nt) { v8f z = {}; acc[nt] = z; }
  float mrow = NEGB;   // running row max (replicated in lanes r and r+16)
  float lrow = 0.f;    // running row sum

#if HAVE_TDM
  const unsigned ldsbase = (unsigned)(unsigned long long)(&hbT[0][0][0]);
  if (wave == 0) tdm_load_tile(hT, ldsbase);          // prefetch tile ks=0
#endif

  for (int ks = 0; ks < NK; ++ks) {
    const int j0 = ks * 32;

#if HAVE_TDM
    const int buf = ks & 1;
    __syncthreads();                     // all waves done reading buf^1
    if (wave == 0) {
      if (ks + 1 < NK) {                 // TDM-prefetch next tile into buf^1
        tdm_load_tile(hT + (size_t)(j0 + 32), ldsbase + (unsigned)((buf ^ 1) * 16384));
        __builtin_amdgcn_s_wait_tensorcnt(1);   // tile ks complete (in-order)
      } else {
        __builtin_amdgcn_s_wait_tensorcnt(0);
      }
    }
    __syncthreads();
#else
    const int buf = 0;
    __syncthreads();
#pragma unroll
    for (int it = 0; it < 2; ++it) {     // plain 2D tile copy (hT is pre-transposed)
      const int c = tid + 128 * it;      // column 0..255
      const v4i* src = (const v4i*)(hT + (size_t)c * N_NODES + j0);
      v4i*       dst = (v4i*)&hbT[0][c][0];
#pragma unroll
      for (int q = 0; q < 4; ++q) dst[q] = src[q];
    }
    __syncthreads();
#endif

    // --- masked e = mish(s_src[i] + s_dst[j]) at A-fragment positions -----
    const int* arow = adj + (size_t)myrow * N_NODES + j0;
    v4i av0 = *(const v4i*)(arow + selA);
    v4i av1 = *(const v4i*)(arow + selA + 4);
    v4i av2 = *(const v4i*)(arow + 16 + selA);
    v4i av3 = *(const v4i*)(arow + 16 + selA + 4);
    if (ks + 1 < NK)
      __builtin_prefetch(arow + 32, 0, 0);       // global_prefetch_b8

    int adjv[16];
#pragma unroll
    for (int q = 0; q < 4; ++q) {
      adjv[q + 0] = av0[q]; adjv[q + 4]  = av1[q];
      adjv[q + 8] = av2[q]; adjv[q + 12] = av3[q];
    }

    float ev[16];
    float tmax = NEGB;
#pragma unroll
    for (int t = 0; t < 16; ++t) {
      const int koff = (t < 8) ? (selA + t) : (8 + selA + t);
      const float sij = ssrc + sdst[j0 + koff];
      const float e   = (adjv[t] > 0) ? mishf(sij) : NEGB;
      ev[t] = e;
      tmax  = fmaxf(tmax, e);
    }
    // each row lives split across lanes (r, r+16): merge halves
    tmax = fmaxf(tmax, __shfl_xor(tmax, 16, 32));
    const bool  grew  = tmax > mrow;
    const float mnew  = fmaxf(mrow, tmax);
    const float scale = __expf(mrow - mnew);

    // --- P fragment (bf16) + row-sum --------------------------------------
    v16bf A;
    float psum = 0.f;
#pragma unroll
    for (int t = 0; t < 16; ++t) {
      const float p = __expf(ev[t] - mnew);
      psum += p;
      A[t] = (__bf16)p;
    }
    psum += __shfl_xor(psum, 16, 32);
    lrow = lrow * scale + psum;
    mrow = mnew;

    // --- rescale accumulators only when some row max actually grew --------
    if (__any(grew)) {
      float sc8[8];
#pragma unroll
      for (int vr = 0; vr < 8; ++vr) sc8[vr] = __shfl(scale, vr + rsel, 32);
#pragma unroll
      for (int nt = 0; nt < 16; ++nt)
#pragma unroll
        for (int vr = 0; vr < 8; ++vr) acc[nt][vr] *= sc8[vr];
    }

    // --- acc += P @ h : 16 WMMA tiles across Fout -------------------------
#pragma unroll
    for (int nt = 0; nt < 16; ++nt) {
      const __bf16* bp = &hbT[buf][nt * 16 + r][selA];
      union { v4i i[2]; v16bf b; } u;
      u.i[0] = *(const v4i*)(bp);       // K selA+0..7   (ds_load_b128)
      u.i[1] = *(const v4i*)(bp + 16);  // K 16+selA+0..7
      acc[nt] = __builtin_amdgcn_wmma_f32_16x16x32_bf16(
          false, A, false, u.b, (short)0, acc[nt], false, false);
    }
  }

  // --- epilogue: out = mish(acc / l) --------------------------------------
  float linv[8];
#pragma unroll
  for (int vr = 0; vr < 8; ++vr)
    linv[vr] = __builtin_amdgcn_rcpf(fmaxf(__shfl(lrow, vr + rsel, 32), 1e-30f));
#pragma unroll
  for (int nt = 0; nt < 16; ++nt)
#pragma unroll
    for (int vr = 0; vr < 8; ++vr) {
      const int row = rowBase + vr + rsel;
      const int col = nt * 16 + r;
      out[(size_t)row * FOUT + col] = mishf(acc[nt][vr] * linv[vr]);
    }
}

// ---------------------------------------------------------------------------
extern "C" void kernel_launch(void* const* d_in, const int* in_sizes, int n_in,
                              void* d_out, int out_size, void* d_ws, size_t ws_size,
                              hipStream_t stream) {
  const float* x   = (const float*)d_in[0];
  const int*   adj = (const int*)  d_in[1];
  const float* w   = (const float*)d_in[2];
  const float* a   = (const float*)d_in[3];

  char* ws = (char*)d_ws;
  float*  h     = (float*)(ws);                         //  8 MB f32 h
  __bf16* hT    = (__bf16*)(ws + 8388608);              //  4 MB bf16 h^T
  float*  s_src = (float*)(ws + 12582912);              // 32 KB
  float*  s_dst = (float*)(ws + 12615680);              // 32 KB
  float*  out   = (float*)d_out;

  gat_gemm1 <<<1024, 256, 0, stream>>>(x, w, h, hT);
  gat_scores<<<1024, 256, 0, stream>>>(h, a, s_src, s_dst);
  gat_flash <<< 128, 128, 0, stream>>>(adj, hT, s_src, s_dst, out);
}
